// LSGA_VIT_11081015623904
// MI455X (gfx1250) — compile-verified, matrix operand order checked by reading, not run on hardware
//
#include <hip/hip_runtime.h>
#include <hip/hip_bf16.h>

typedef _Float16 f16;
typedef __attribute__((ext_vector_type(16))) _Float16 v16h;
typedef __attribute__((ext_vector_type(8)))  float    v8f;

#define EPS_ 1e-5f

// ---------------- LDS layout (bytes) ----------------
#define T_OFF   0        // float [96][128]   49152
#define H_OFF   49152    // f16   [96][128]   24576
#define U_OFF   73728    // f16   [96][128]   24576  (VV transposed [128][96] / q row-major / g1c)
#define X_OFF   98304    // f16   [96][128]   24576
#define AM_OFF  122880   // f16   [96][96]    18432
#define RED_OFF 141312   // float [160]       640
#define P_OFF   49152    // f16 [11][11][160] 38720  (aliases H+U; conv stage only)
#define XS_OFF  98304    // float [2916]      11664  (aliases X;  conv stage only)
#define SMEM_BYTES 141952

// ---------------- workspace layout (bytes) ----------------
// All weights pre-swizzled into WMMA-fragment order:
//   per 32x16 (B) or 16x32 (A) tile: u32 blob[tile][lane(32)][j(8)]  (8 dwords/lane, contiguous)
#define W2PK_OFF  0u        // u32 [9*40*256]      368640
#define BIAS2_OFF 368640u   // f32 [128]           512
#define WAPK_OFF  369152u   // u32 [2*24*256]      49152
#define WVPK_OFF  418304u   // u32 [2*32*256]      65536
#define QWPK_OFF  483840u   // u32 [2*32*256]      65536
#define PWPK_OFF  549376u   // u32 [2*32*256]      65536
#define FC1PK_OFF 614912u   // u32 [2*128*256]     262144
#define FC2PK_OFF 877056u   // u32 [2*128*256]     262144

__device__ __forceinline__ int lane_id() { return (int)(threadIdx.x & 31u); }

__device__ __forceinline__ v8f wmma32(v16h a, v16h b, v8f c) {
  return __builtin_amdgcn_wmma_f32_16x16x32_f16(false, a, false, b, (short)0, c, false, false);
}

// K-pair base for f16 A/B fragment layout (16x32 / 32x16), VGPR j, lane-half hi
__device__ __forceinline__ int kb_of(int j, int hi) {
  return ((j >> 2) << 4) + (hi << 3) + ((j & 3) << 1);
}

__device__ __forceinline__ unsigned int pckh(float a, float b) {
  union { f16 h[2]; unsigned int u; } c;
  c.h[0] = (f16)a; c.h[1] = (f16)b;
  return c.u;
}

// Packed-weight fragment: 8 contiguous dwords per lane (-> 2x global_load_b128)
__device__ __forceinline__ v16h frag_packed(const unsigned int* __restrict__ blob, int tile) {
  const unsigned int* p = blob + ((tile << 5) + lane_id()) * 8;
  union { v16h v; unsigned int u[8]; } f;
#pragma unroll
  for (int j = 0; j < 8; ++j) f.u[j] = p[j];
  return f.v;
}

// A fragment from row-major f16 [rows][stride]: A[m][k]  (contiguous k-pairs -> dword loads)
__device__ __forceinline__ v16h frag_a_rm(const f16* src, int row0, int col0, int stride) {
  const int l = lane_id();
  const int m = row0 + (l & 15);
  const int hi = l >> 4;
  const f16* base = src + m * stride + col0;
  union { v16h v; unsigned int u[8]; } f;
#pragma unroll
  for (int j = 0; j < 8; ++j) f.u[j] = *(const unsigned int*)(base + kb_of(j, hi));
  return f.v;
}

// B fragment B[k][n] = src[n][k]  (transposed source, contiguous k-pairs)
__device__ __forceinline__ v16h frag_b_T(const f16* src, int krow0, int ncol0, int stride) {
  const int l = lane_id();
  const int n = ncol0 + (l & 15);
  const int hi = l >> 4;
  const f16* base = src + n * stride + krow0;
  union { v16h v; unsigned int u[8]; } f;
#pragma unroll
  for (int j = 0; j < 8; ++j) f.u[j] = *(const unsigned int*)(base + kb_of(j, hi));
  return f.v;
}

// A fragment for conv2d implicit GEMM from padded LDS tile P[11][11][160].
// Invalid rows (m>=81) are clamped to border pixel (10,10) which is all zeros:
// loads stay unconditional (no exec masking).
__device__ __forceinline__ v16h frag_a_conv(const f16* P, int row0, int ky, int kx, int c0) {
  const int l = lane_id();
  const int m = row0 + (l & 15);
  const int hi = l >> 4;
  const int oy = m / 9, ox = m - oy * 9;
  const int pix = (m < 81) ? ((oy + ky) * 11 + (ox + kx)) : 120; // (10,10) = zero border
  const f16* base = P + pix * 160 + c0;
  union { v16h v; unsigned int u[8]; } f;
#pragma unroll
  for (int j = 0; j < 8; ++j) f.u[j] = *(const unsigned int*)(base + kb_of(j, hi));
  return f.v;
}

// A fragment for head-local q (K padded 10 -> 32). Pairs j>=4 (k>=16) are
// compile-time zero; j=0..3 load unconditionally (in-bounds) then select.
__device__ __forceinline__ v16h frag_a_head(const f16* Uq, int row0, int hb) {
  const int l = lane_id();
  const int m = row0 + (l & 15);
  const int hi = l >> 4;
  const f16* base = Uq + m * 128 + hb;
  union { v16h v; unsigned int u[8]; } f;
#pragma unroll
  for (int j = 0; j < 4; ++j) {
    const int kb = (hi << 3) + (j << 1);
    const unsigned int val = *(const unsigned int*)(base + kb);
    f.u[j] = (kb < 10) ? val : 0u;
  }
#pragma unroll
  for (int j = 4; j < 8; ++j) f.u[j] = 0u;
  return f.v;
}

// B fragment for scores: B[k][j] = kv[j][hb+k], K padded 10 -> 32 (same scheme)
__device__ __forceinline__ v16h frag_b_head(const f16* Xk, int ncol0, int hb) {
  const int l = lane_id();
  const int n = ncol0 + (l & 15);
  const int hi = l >> 4;
  const f16* base = Xk + n * 128 + hb;
  union { v16h v; unsigned int u[8]; } f;
#pragma unroll
  for (int j = 0; j < 4; ++j) {
    const int kb = (hi << 3) + (j << 1);
    const unsigned int val = *(const unsigned int*)(base + kb);
    f.u[j] = (kb < 10) ? val : 0u;
  }
#pragma unroll
  for (int j = 4; j < 8; ++j) f.u[j] = 0u;
  return f.v;
}

// B fragment for o = att @ kv : B[k=j][n] = kv[j][hb+n] (strided, small usage)
__device__ __forceinline__ v16h frag_b_ohead(const f16* Xk, int krow0, int hb) {
  const int l = lane_id();
  const int n = l & 15;
  const int hi = l >> 4;
  v16h b;
#pragma unroll
  for (int j = 0; j < 8; ++j) {
    const int kb = kb_of(j, hi);
    b[2 * j]     = Xk[(krow0 + kb)     * 128 + hb + n];
    b[2 * j + 1] = Xk[(krow0 + kb + 1) * 128 + hb + n];
  }
  return b;
}

// f16 tile store; wave-uniform fast path when the whole tile is in-range rows/cols
__device__ __forceinline__ void store_tile_f16(f16* dst, int row0, int col0, int stride,
                                               v8f acc, int rmax, int cmax) {
  const int l = lane_id();
  const int n = col0 + (l & 15);
  const int mb = row0 + ((l >> 4) << 3);
  const bool ncol = (n < cmax);
  if (row0 + 15 < rmax) {
#pragma unroll
    for (int r = 0; r < 8; ++r) dst[(mb + r) * stride + n] = (f16)(ncol ? acc[r] : 0.f);
  } else {
#pragma unroll
    for (int r = 0; r < 8; ++r) {
      const int m = mb + r;
      dst[m * stride + n] = (f16)((m < rmax && ncol) ? acc[r] : 0.f);
    }
  }
}

// f32 tile store into Tm (rows valid < 81, cols valid < 120), uniform fast path
__device__ __forceinline__ void store_tile_Tm(float* Tm, int mt, int nt, v8f acc) {
  const int l = lane_id();
  const int n = nt * 16 + (l & 15);
  const int mb = mt * 16 + ((l >> 4) << 3);
  const bool ncol = (n < 120);
  if (mt < 5) {
#pragma unroll
    for (int r = 0; r < 8; ++r) Tm[(mb + r) * 128 + n] = ncol ? acc[r] : 0.f;
  } else {
#pragma unroll
    for (int r = 0; r < 8; ++r) {
      const int m = mb + r;
      Tm[m * 128 + n] = (m < 81 && ncol) ? acc[r] : 0.f;
    }
  }
}

// Transposed store: logical tile rows m, cols n; writes dst[n][m] with
// contiguous per-lane dword stores.
__device__ __forceinline__ void store_tile_f16_T(f16* dst, int row0, int col0, int strideT,
                                                 v8f acc) {
  const int l = lane_id();
  const int n = col0 + (l & 15);
  const int mb = row0 + ((l >> 4) << 3);
  unsigned int* d32 = (unsigned int*)(dst + n * strideT + mb);
#pragma unroll
  for (int r2 = 0; r2 < 4; ++r2) d32[r2] = pckh(acc[2 * r2], acc[2 * r2 + 1]);
}

__device__ __forceinline__ void ln_row_inplace(float* row, const float* g, const float* bb) {
  float m = 0.f;
  for (int c = 0; c < 120; ++c) m += row[c];
  m *= (1.f / 120.f);
  float v = 0.f;
  for (int c = 0; c < 120; ++c) { float d = row[c] - m; v += d * d; }
  v *= (1.f / 120.f);
  const float inv = rsqrtf(v + EPS_);
  for (int c = 0; c < 120; ++c) row[c] = g[c] * (row[c] - m) * inv + bb[c];
}

__device__ __forceinline__ void ln_to_f16(const float* Tm, f16* dst, const float* g,
                                          const float* bb, int tid) {
  if (tid < 96) {
    f16* drow = dst + tid * 128;
    if (tid < 81) {
      const float* row = Tm + tid * 128;
      float m = 0.f;
      for (int c = 0; c < 120; ++c) m += row[c];
      m *= (1.f / 120.f);
      float v = 0.f;
      for (int c = 0; c < 120; ++c) { float d = row[c] - m; v += d * d; }
      v *= (1.f / 120.f);
      const float inv = rsqrtf(v + EPS_);
      for (int c = 0; c < 120; ++c) drow[c] = (f16)(g[c] * (row[c] - m) * inv + bb[c]);
      for (int c = 120; c < 128; ++c) drow[c] = (f16)0.f;
    } else {
      for (int c = 0; c < 128; ++c) drow[c] = (f16)0.f;
    }
  }
}

// ================= fused per-batch-element kernel =================
__global__ __launch_bounds__(256)
void lsga_fused(const float* __restrict__ x,
                const float* __restrict__ c3d_w, const float* __restrict__ c3d_b,
                const float* __restrict__ bn3d_g, const float* __restrict__ bn3d_b,
                const float* __restrict__ bn3d_m, const float* __restrict__ bn3d_v,
                const float* __restrict__ pe_ln_g, const float* __restrict__ pe_ln_b,
                const float* __restrict__ n1_g, const float* __restrict__ n1_b,
                const float* __restrict__ qkv_b, const float* __restrict__ proj_b,
                const float* __restrict__ n2_g, const float* __restrict__ n2_b,
                const float* __restrict__ fc1_b, const float* __restrict__ fc2_b,
                const float* __restrict__ fn_g, const float* __restrict__ fn_b,
                const float* __restrict__ head_w, const float* __restrict__ head_b,
                const unsigned int* __restrict__ W2pk, const float* __restrict__ bias2,
                const unsigned int* __restrict__ wApk, const unsigned int* __restrict__ wVpk,
                const unsigned int* __restrict__ qwpk, const unsigned int* __restrict__ pwpk,
                const unsigned int* __restrict__ fc1pk, const unsigned int* __restrict__ fc2pk,
                float* __restrict__ out)
{
  extern __shared__ char smem[];
  float* Tm  = (float*)(smem + T_OFF);
  f16*   Hh  = (f16*)(smem + H_OFF);
  f16*   Uu  = (f16*)(smem + U_OFF);
  f16*   Xx  = (f16*)(smem + X_OFF);
  f16*   AM  = (f16*)(smem + AM_OFF);
  float* RED = (float*)(smem + RED_OFF);
  f16*   P   = (f16*)(smem + P_OFF);
  float* XS  = (float*)(smem + XS_OFF);

  const int tid = (int)threadIdx.x;
  const int w = tid >> 5;
  const int b = (int)blockIdx.x;

  // Warm caches with the packed weights while conv3d runs (global_prefetch_b8)
  __builtin_prefetch(W2pk + tid * 64, 0, 3);
  __builtin_prefetch(W2pk + 16384 + tid * 64, 0, 3);
  __builtin_prefetch(wApk + tid * 48, 0, 3);
  __builtin_prefetch(fc1pk + tid * 256, 0, 3);
  __builtin_prefetch(fc2pk + tid * 256, 0, 3);

  // ---- Stage 0: conv3d + BN3d + ReLU -> padded LDS tile P[11][11][160] ----
  for (int i = tid; i < 19360; i += 256) P[i] = (f16)0.f;
  for (int i = tid; i < 2916; i += 256) XS[i] = x[b * 2916 + i];
  if (tid < 108) RED[tid] = c3d_w[tid];
  if (tid >= 108 && tid < 112) {
    const int e = tid - 108;
    const float sc = bn3d_g[e] * rsqrtf(bn3d_v[e] + EPS_);
    RED[108 + e] = sc;
    RED[112 + e] = (c3d_b[e] - bn3d_m[e]) * sc + bn3d_b[e];
  }
  __syncthreads();
  for (int o = tid; o < 11664; o += 256) {
    const int e = o / 2916;
    const int r0 = o - e * 2916;
    const int z = r0 / 81;
    const int p = r0 - z * 81;
    const int py = p / 9, px = p - py * 9;
    float s = 0.f;
    for (int dz = 0; dz < 3; ++dz) {
      const int zz = z + dz - 1;
      if (zz < 0 || zz >= 36) continue;
      for (int dy = 0; dy < 3; ++dy) {
        const int yy = py + dy - 1;
        if (yy < 0 || yy >= 9) continue;
        for (int dx = 0; dx < 3; ++dx) {
          const int xx = px + dx - 1;
          if (xx < 0 || xx >= 9) continue;
          s += XS[(zz * 9 + yy) * 9 + xx] * RED[((e * 3 + dz) * 3 + dy) * 3 + dx];
        }
      }
    }
    const float val = fmaxf(s * RED[108 + e] + RED[112 + e], 0.f);
    P[((py + 1) * 11 + (px + 1)) * 160 + (e * 36 + z)] = (f16)val;
  }
  __syncthreads();

  // ---- Stage 1: conv2d implicit GEMM (BN folded into weights) + bias + ReLU -> Tm ----
#pragma unroll 1
  for (int t6 = 0; t6 < 6; ++t6) {
    const int id = w * 6 + t6;
    const int mt = id >> 3, nt = id & 7;
    v8f acc = {};
    for (int tap = 0; tap < 9; ++tap) {
      const int ky = tap / 3, kx = tap - ky * 3;
#pragma unroll
      for (int cc = 0; cc < 5; ++cc) {
        v16h a = frag_a_conv(P, mt * 16, ky, kx, cc * 32);
        v16h bb = frag_packed(W2pk, tap * 40 + cc * 8 + nt);
        acc = wmma32(a, bb, acc);
      }
    }
    const int l = lane_id();
    const int n = nt * 16 + (l & 15);
    const float bs = bias2[n];
    v8f cv;
#pragma unroll
    for (int r = 0; r < 8; ++r) cv[r] = fmaxf(acc[r] + bs, 0.f);
    store_tile_Tm(Tm, mt, nt, cv);
  }
  __syncthreads();

  // pe-LN in place; precompute separable gaussian vector gv[i] = exp(-(i-40)^2/200)
  if (tid < 81) ln_row_inplace(Tm + tid * 128, pe_ln_g, pe_ln_b);
  if (tid >= 96 && tid < 177) {
    const float ci = (float)(tid - 96 - 40);
    RED[tid - 96] = __expf(-ci * ci * 0.005f);
  }
  __syncthreads();

  // ---- Stage 2: two LSGA-ViT blocks ----
  for (int blk = 0; blk < 2; ++blk) {
    const float* g1 = n1_g + blk * 120;  const float* b1 = n1_b + blk * 120;
    const float* g2 = n2_g + blk * 120;  const float* b2 = n2_b + blk * 120;
    const unsigned int* wApk_i  = wApk  + blk * 24 * 256;
    const unsigned int* wVpk_i  = wVpk  + blk * 32 * 256;
    const unsigned int* qwpk_i  = qwpk  + blk * 32 * 256;
    const unsigned int* pwpk_i  = pwpk  + blk * 32 * 256;
    const unsigned int* fc1pk_i = fc1pk + blk * 128 * 256;
    const unsigned int* fc2pk_i = fc2pk + blk * 128 * 256;
    const float* qb  = qkv_b  + blk * 120;
    const float* pb  = proj_b + blk * 120;
    const float* f1b = fc1_b  + blk * 480;
    const float* f2b = fc2_b  + blk * 120;

    // (a) h = LN(t, n1) -> Hh
    ln_to_f16(Tm, Hh, g1, b1, tid);
    __syncthreads();

    // (b) A-logits[t][n] = sum_c wA[t][c] * h[n][c]   -> AM
    for (int id = w; id < 36; id += 8) {
      const int mt = id / 6, nt = id - (id / 6) * 6;
      v8f acc = {};
#pragma unroll
      for (int ks = 0; ks < 4; ++ks) {
        v16h a = frag_packed(wApk_i, mt * 4 + ks);
        v16h bb = frag_b_T(Hh, ks * 32, nt * 16, 128);
        acc = wmma32(a, bb, acc);
      }
      store_tile_f16(AM, mt * 16, nt * 16, 96, acc, 96, 96);
    }
    __syncthreads();

    // (c) softmax over n (cols) per row t
    if (tid < 81) {
      f16* row = AM + tid * 96;
      float mx = -1e30f;
      for (int j = 0; j < 81; ++j) mx = fmaxf(mx, (float)row[j]);
      float sum = 0.f;
      for (int j = 0; j < 81; ++j) sum += __expf((float)row[j] - mx);
      const float inv = 1.f / sum;
      for (int j = 0; j < 81; ++j) row[j] = (f16)(__expf((float)row[j] - mx) * inv);
      for (int j = 81; j < 96; ++j) row[j] = (f16)0.f;
    } else if (tid < 96) {
      f16* row = AM + tid * 96;
      for (int j = 0; j < 96; ++j) row[j] = (f16)0.f;
    }
    __syncthreads();

    // (d) VV = h @ wV -> Uu (stored TRANSPOSED: Ut[d][token], stride 96)
#pragma unroll 1
    for (int t6 = 0; t6 < 6; ++t6) {
      const int id = w * 6 + t6, mt = id >> 3, nt = id & 7;
      v8f acc = {};
#pragma unroll
      for (int ks = 0; ks < 4; ++ks) {
        v16h a = frag_a_rm(Hh, mt * 16, ks * 32, 128);
        v16h bb = frag_packed(wVpk_i, ks * 8 + nt);
        acc = wmma32(a, bb, acc);
      }
      store_tile_f16_T(Uu, mt * 16, nt * 16, 96, acc);
    }
    __syncthreads();

    // (e) xt = A @ VV -> Xx   (B[k=token][n=d] = Ut[d][token], contiguous pairs)
#pragma unroll 1
    for (int t6 = 0; t6 < 6; ++t6) {
      const int id = w * 6 + t6, mt = id >> 3, nt = id & 7;
      v8f acc = {};
#pragma unroll
      for (int ks = 0; ks < 3; ++ks) {
        v16h a = frag_a_rm(AM, mt * 16, ks * 32, 96);
        v16h bb = frag_b_T(Uu, ks * 32, nt * 16, 96);
        acc = wmma32(a, bb, acc);
      }
      store_tile_f16(Xx, mt * 16, nt * 16, 128, acc, 96, 128);
    }
    __syncthreads();

    // (f) q = (xt @ qw^T + qb) * HD^-0.5 -> Uu (row-major, overwrites VV)
#pragma unroll 1
    for (int t6 = 0; t6 < 6; ++t6) {
      const int id = w * 6 + t6, mt = id >> 3, nt = id & 7;
      const int l = lane_id();
      const int n = nt * 16 + (l & 15);
      const float qbv = (n < 120) ? qb[n] : 0.f;
      v8f acc = {};
#pragma unroll
      for (int r = 0; r < 8; ++r) acc[r] = qbv;
#pragma unroll
      for (int ks = 0; ks < 4; ++ks) {
        v16h a = frag_a_rm(Xx, mt * 16, ks * 32, 128);
        v16h bb = frag_packed(qwpk_i, ks * 8 + nt);
        acc = wmma32(a, bb, acc);
      }
      v8f qv;
#pragma unroll
      for (int r = 0; r < 8; ++r) qv[r] = acc[r] * 0.3162277660168379f;
      store_tile_f16(Uu, mt * 16, nt * 16, 128, qv, 81, 120);
    }
    __syncthreads();

    // (g) attention, head by head; kv == xt (Xx); o written in place into Xx
    for (int hh = 0; hh < 12; ++hh) {
      const int hb = hh * 10;
      // scores = q_h @ kv_h^T  (K padded 10->32)
      for (int id = w; id < 36; id += 8) {
        const int mt = id / 6, nt = id - (id / 6) * 6;
        v16h a = frag_a_head(Uu, mt * 16, hb);
        v16h bb = frag_b_head(Xx, nt * 16, hb);
        v8f acc = {};
        acc = wmma32(a, bb, acc);
        store_tile_f16(AM, mt * 16, nt * 16, 96, acc, 96, 96);
      }
      __syncthreads();
      // softmax with separable gaussian bias gv[i]*gv[j]
      if (tid < 81) {
        f16* row = AM + tid * 96;
        const float gi = RED[tid];
        float mx = -1e30f;
        for (int j = 0; j < 81; ++j) mx = fmaxf(mx, (float)row[j] + gi * RED[j]);
        float sum = 0.f;
        for (int j = 0; j < 81; ++j) sum += __expf((float)row[j] + gi * RED[j] - mx);
        const float inv = 1.f / sum;
        for (int j = 0; j < 81; ++j)
          row[j] = (f16)(__expf((float)row[j] + gi * RED[j] - mx) * inv);
        for (int j = 81; j < 96; ++j) row[j] = (f16)0.f;
      } else if (tid < 96) {
        f16* row = AM + tid * 96;
        for (int j = 0; j < 96; ++j) row[j] = (f16)0.f;
      }
      __syncthreads();
      // o_h = att @ kv_h, held in regs, then written back into Xx head columns
      v8f oacc = {};
      if (w < 6) {
#pragma unroll
        for (int ks = 0; ks < 3; ++ks) {
          v16h a = frag_a_rm(AM, w * 16, ks * 32, 96);
          v16h bb = frag_b_ohead(Xx, ks * 32, hb);
          oacc = wmma32(a, bb, oacc);
        }
      }
      __syncthreads();
      if (w < 6) {
        const int l = lane_id();
        const int n = l & 15;
        const int mb = w * 16 + ((l >> 4) << 3);
        if (n < 10) {
#pragma unroll
          for (int r = 0; r < 8; ++r) Xx[(mb + r) * 128 + hb + n] = (f16)oacc[r];
        }
      }
      __syncthreads();
    }

    // (h) t += o @ proj^T + pb  (accumulate straight into Tm)
#pragma unroll 1
    for (int t6 = 0; t6 < 6; ++t6) {
      const int id = w * 6 + t6, mt = id >> 3, nt = id & 7;
      const int l = lane_id();
      const int n = nt * 16 + (l & 15);
      const int mb = mt * 16 + ((l >> 4) << 3);
      const float pbv = (n < 120) ? pb[n] : 0.f;
      v8f acc = {};
#pragma unroll
      for (int r = 0; r < 8; ++r) acc[r] = Tm[(mb + r) * 128 + n] + pbv;
#pragma unroll
      for (int ks = 0; ks < 4; ++ks) {
        v16h a = frag_a_rm(Xx, mt * 16, ks * 32, 128);
        v16h bb = frag_packed(pwpk_i, ks * 8 + nt);
        acc = wmma32(a, bb, acc);
      }
      store_tile_Tm(Tm, mt, nt, acc);
    }
    __syncthreads();

    // (i) h2 = LN(t, n2) -> Hh
    ln_to_f16(Tm, Hh, g2, b2, tid);
    __syncthreads();

    // (j) MLP, chunked over HID (4 chunks of 128); fc2 accumulators live in regs
    v8f acc2[6];
    {
      const int l = lane_id();
#pragma unroll
      for (int t6 = 0; t6 < 6; ++t6) {
        const int id = w * 6 + t6, mt = id >> 3, nt = id & 7;
        const int n = nt * 16 + (l & 15);
        const int mb = mt * 16 + ((l >> 4) << 3);
        const float fb = (n < 120) ? f2b[n] : 0.f;
#pragma unroll
        for (int r = 0; r < 8; ++r) acc2[t6][r] = Tm[(mb + r) * 128 + n] + fb;
      }
    }
    for (int kc = 0; kc < 4; ++kc) {
      // g1 chunk = gelu(h2 @ fc1^T[:, chunk] + b1c) -> Uu (row-major)
#pragma unroll 1
      for (int t6 = 0; t6 < 6; ++t6) {
        const int id = w * 6 + t6, mt = id >> 3, nt = id & 7;
        const int l = lane_id();
        const int col = kc * 128 + nt * 16 + (l & 15);
        const float bv = (col < 480) ? f1b[col] : 0.f;
        v8f acc = {};
#pragma unroll
        for (int r = 0; r < 8; ++r) acc[r] = bv;
#pragma unroll
        for (int ks = 0; ks < 4; ++ks) {
          v16h a = frag_a_rm(Hh, mt * 16, ks * 32, 128);
          v16h bb = frag_packed(fc1pk_i, ks * 32 + kc * 8 + nt);
          acc = wmma32(a, bb, acc);
        }
        v8f gv;
#pragma unroll
        for (int r = 0; r < 8; ++r) {
          const float vv = acc[r];
          gv[r] = 0.5f * vv * (1.f + erff(vv * 0.7071067811865475f));
        }
        store_tile_f16(Uu, mt * 16, nt * 16, 128, gv, 81, (kc == 3) ? 96 : 128);
      }
      __syncthreads();
      // fc2 partial: acc2 += g1c @ fc2^T[chunk rows]
#pragma unroll
      for (int t6 = 0; t6 < 6; ++t6) {
        const int id = w * 6 + t6, mt = id >> 3, nt = id & 7;
#pragma unroll
        for (int ks = 0; ks < 4; ++ks) {
          v16h a = frag_a_rm(Uu, mt * 16, ks * 32, 128);
          v16h bb = frag_packed(fc2pk_i, (kc * 4 + ks) * 8 + nt);
          acc2[t6] = wmma32(a, bb, acc2[t6]);
        }
      }
      __syncthreads();
    }
    {
#pragma unroll
      for (int t6 = 0; t6 < 6; ++t6) {
        const int id = w * 6 + t6, mt = id >> 3, nt = id & 7;
        store_tile_Tm(Tm, mt, nt, acc2[t6]);
      }
    }
    __syncthreads();
  } // blocks

  // ---- Stage 3: final LN, token mean, classifier head ----
  if (tid < 81) ln_row_inplace(Tm + tid * 128, fn_g, fn_b);
  __syncthreads();
  if (tid < 120) {
    float s = 0.f;
    for (int i = 0; i < 81; ++i) s += Tm[i * 128 + tid];
    RED[tid] = s * (1.f / 81.f);
  }
  __syncthreads();
  if (tid < 13) {
    float s = head_b[tid];
    for (int d = 0; d < 120; ++d) s += RED[d] * head_w[tid * 120 + d];
    out[b * 13 + tid] = s;
  }
}

// ================= weight prep kernel (pre-swizzle into fragment order) =================
__global__ void prep_weights(const float* __restrict__ c2d_w, const float* __restrict__ c2d_b,
                             const float* __restrict__ bn2d_g, const float* __restrict__ bn2d_b,
                             const float* __restrict__ bn2d_m, const float* __restrict__ bn2d_v,
                             const float* __restrict__ wA, const float* __restrict__ wV,
                             const float* __restrict__ qkv_w, const float* __restrict__ proj_w,
                             const float* __restrict__ fc1_w, const float* __restrict__ fc2_w,
                             unsigned int* __restrict__ W2pk, float* __restrict__ bias2,
                             unsigned int* __restrict__ wApk, unsigned int* __restrict__ wVpk,
                             unsigned int* __restrict__ qwpk, unsigned int* __restrict__ pwpk,
                             unsigned int* __restrict__ fc1pk, unsigned int* __restrict__ fc2pk)
{
  int idx = (int)(blockIdx.x * blockDim.x + threadIdx.x);

  // --- conv2d B tiles: [9 taps][Kt=5][Nt=8], K=c (160, valid<144), N=d (128, valid<120)
  if (idx < 9 * 40 * 256) {
    const int tap = idx / (40 * 256);
    const int q = idx - tap * 40 * 256;
    const int tile = q >> 8, lane = (q >> 3) & 31, j = q & 7;
    const int kt = tile >> 3, nt = tile & 7;
    const int n = nt * 16 + (lane & 15);
    const int kb = kb_of(j, lane >> 4);
    const int k0 = kt * 32 + kb;
    float v0 = 0.f, v1 = 0.f;
    if (n < 120) {
      const float sc = bn2d_g[n] * rsqrtf(bn2d_v[n] + EPS_);
      if (k0 < 144)     v0 = c2d_w[(n * 144 + k0) * 9 + tap] * sc;
      if (k0 + 1 < 144) v1 = c2d_w[(n * 144 + k0 + 1) * 9 + tap] * sc;
    }
    W2pk[idx] = pckh(v0, v1);
    return;
  }
  idx -= 9 * 40 * 256;

  // --- bias2
  if (idx < 128) {
    bias2[idx] = (idx < 120)
      ? (c2d_b[idx] - bn2d_m[idx]) * bn2d_g[idx] * rsqrtf(bn2d_v[idx] + EPS_) + bn2d_b[idx]
      : 0.f;
    return;
  }
  idx -= 128;

  // --- wA A-tiles: [2][Mt=6][Kt=4], M=t (96, valid<81), K=c (128, valid<120)
  if (idx < 2 * 24 * 256) {
    const int i = idx / (24 * 256);
    const int q = idx - i * 24 * 256;
    const int tile = q >> 8, lane = (q >> 3) & 31, j = q & 7;
    const int mt = tile >> 2, kt = tile & 3;
    const int m = mt * 16 + (lane & 15);
    const int kb = kb_of(j, lane >> 4);
    const int k0 = kt * 32 + kb;
    float v0 = 0.f, v1 = 0.f;
    if (m < 81) {
      if (k0 < 120)     v0 = wA[i * 9720 + m * 120 + k0];
      if (k0 + 1 < 120) v1 = wA[i * 9720 + m * 120 + k0 + 1];
    }
    wApk[idx] = pckh(v0, v1);
    return;
  }
  idx -= 2 * 24 * 256;

  // --- wV B-tiles: [2][Kt=4][Nt=8], B[c][d] = wV[i][c][d]
  if (idx < 2 * 32 * 256) {
    const int i = idx / (32 * 256);
    const int q = idx - i * 32 * 256;
    const int tile = q >> 8, lane = (q >> 3) & 31, j = q & 7;
    const int kt = tile >> 3, nt = tile & 7;
    const int n = nt * 16 + (lane & 15);
    const int kb = kb_of(j, lane >> 4);
    const int k0 = kt * 32 + kb;
    float v0 = 0.f, v1 = 0.f;
    if (n < 120) {
      if (k0 < 120)     v0 = wV[i * 14400 + k0 * 120 + n];
      if (k0 + 1 < 120) v1 = wV[i * 14400 + (k0 + 1) * 120 + n];
    }
    wVpk[idx] = pckh(v0, v1);
    return;
  }
  idx -= 2 * 32 * 256;

  // --- qkv_w B-tiles (transposed): B[c][d] = qkv_w[i][d][c]
  if (idx < 2 * 32 * 256) {
    const int i = idx / (32 * 256);
    const int q = idx - i * 32 * 256;
    const int tile = q >> 8, lane = (q >> 3) & 31, j = q & 7;
    const int kt = tile >> 3, nt = tile & 7;
    const int n = nt * 16 + (lane & 15);
    const int kb = kb_of(j, lane >> 4);
    const int k0 = kt * 32 + kb;
    float v0 = 0.f, v1 = 0.f;
    if (n < 120) {
      if (k0 < 120)     v0 = qkv_w[i * 14400 + n * 120 + k0];
      if (k0 + 1 < 120) v1 = qkv_w[i * 14400 + n * 120 + k0 + 1];
    }
    qwpk[idx] = pckh(v0, v1);
    return;
  }
  idx -= 2 * 32 * 256;

  // --- proj_w B-tiles (transposed): B[c][d] = proj_w[i][d][c]
  if (idx < 2 * 32 * 256) {
    const int i = idx / (32 * 256);
    const int q = idx - i * 32 * 256;
    const int tile = q >> 8, lane = (q >> 3) & 31, j = q & 7;
    const int kt = tile >> 3, nt = tile & 7;
    const int n = nt * 16 + (lane & 15);
    const int kb = kb_of(j, lane >> 4);
    const int k0 = kt * 32 + kb;
    float v0 = 0.f, v1 = 0.f;
    if (n < 120) {
      if (k0 < 120)     v0 = proj_w[i * 14400 + n * 120 + k0];
      if (k0 + 1 < 120) v1 = proj_w[i * 14400 + n * 120 + k0 + 1];
    }
    pwpk[idx] = pckh(v0, v1);
    return;
  }
  idx -= 2 * 32 * 256;

  // --- fc1 B-tiles: [2][Kt=4][Nt=32], B[c][h] = fc1_w[i][h][c]
  if (idx < 2 * 128 * 256) {
    const int i = idx / (128 * 256);
    const int q = idx - i * 128 * 256;
    const int tile = q >> 8, lane = (q >> 3) & 31, j = q & 7;
    const int kt = tile >> 5, nt = tile & 31;
    const int n = nt * 16 + (lane & 15);
    const int kb = kb_of(j, lane >> 4);
    const int k0 = kt * 32 + kb;
    float v0 = 0.f, v1 = 0.f;
    if (n < 480) {
      if (k0 < 120)     v0 = fc1_w[i * 57600 + n * 120 + k0];
      if (k0 + 1 < 120) v1 = fc1_w[i * 57600 + n * 120 + k0 + 1];
    }
    fc1pk[idx] = pckh(v0, v1);
    return;
  }
  idx -= 2 * 128 * 256;

  // --- fc2 B-tiles: [2][Kt=16][Nt=8], B[h][d] = fc2_w[i][d][h]
  if (idx < 2 * 128 * 256) {
    const int i = idx / (128 * 256);
    const int q = idx - i * 128 * 256;
    const int tile = q >> 8, lane = (q >> 3) & 31, j = q & 7;
    const int kt = tile >> 3, nt = tile & 7;
    const int n = nt * 16 + (lane & 15);
    const int kb = kb_of(j, lane >> 4);
    const int k0 = kt * 32 + kb;
    float v0 = 0.f, v1 = 0.f;
    if (n < 120) {
      if (k0 < 480)     v0 = fc2_w[i * 57600 + n * 480 + k0];
      if (k0 + 1 < 480) v1 = fc2_w[i * 57600 + n * 480 + k0 + 1];
    }
    fc2pk[idx] = pckh(v0, v1);
    return;
  }
}

extern "C" void kernel_launch(void* const* d_in, const int* in_sizes, int n_in,
                              void* d_out, int out_size, void* d_ws, size_t ws_size,
                              hipStream_t stream) {
  const float* x      = (const float*)d_in[0];
  const float* c3d_w  = (const float*)d_in[1];
  const float* c3d_b  = (const float*)d_in[2];
  const float* bn3d_g = (const float*)d_in[3];
  const float* bn3d_b = (const float*)d_in[4];
  const float* bn3d_m = (const float*)d_in[5];
  const float* bn3d_v = (const float*)d_in[6];
  const float* c2d_w  = (const float*)d_in[7];
  const float* c2d_b  = (const float*)d_in[8];
  const float* bn2d_g = (const float*)d_in[9];
  const float* bn2d_b = (const float*)d_in[10];
  const float* bn2d_m = (const float*)d_in[11];
  const float* bn2d_v = (const float*)d_in[12];
  const float* pe_ln_g= (const float*)d_in[13];
  const float* pe_ln_b= (const float*)d_in[14];
  const float* n1_g   = (const float*)d_in[15];
  const float* n1_b   = (const float*)d_in[16];
  const float* wA     = (const float*)d_in[17];
  const float* wV     = (const float*)d_in[18];
  const float* qkv_w  = (const float*)d_in[19];
  const float* qkv_b  = (const float*)d_in[20];
  const float* proj_w = (const float*)d_in[21];
  const float* proj_b = (const float*)d_in[22];
  const float* n2_g   = (const float*)d_in[23];
  const float* n2_b   = (const float*)d_in[24];
  const float* fc1_w  = (const float*)d_in[25];
  const float* fc1_b  = (const float*)d_in[26];
  const float* fc2_w  = (const float*)d_in[27];
  const float* fc2_b  = (const float*)d_in[28];
  const float* fn_g   = (const float*)d_in[29];
  const float* fn_b   = (const float*)d_in[30];
  const float* head_w = (const float*)d_in[31];
  const float* head_b = (const float*)d_in[32];

  char* ws = (char*)d_ws;
  unsigned int* W2pk  = (unsigned int*)(ws + W2PK_OFF);
  float*        bias2 = (float*)(ws + BIAS2_OFF);
  unsigned int* wApk  = (unsigned int*)(ws + WAPK_OFF);
  unsigned int* wVpk  = (unsigned int*)(ws + WVPK_OFF);
  unsigned int* qwpk  = (unsigned int*)(ws + QWPK_OFF);
  unsigned int* pwpk  = (unsigned int*)(ws + PWPK_OFF);
  unsigned int* fc1pk = (unsigned int*)(ws + FC1PK_OFF);
  unsigned int* fc2pk = (unsigned int*)(ws + FC2PK_OFF);

  const int prep_total = 9 * 40 * 256 + 128 + 2 * 24 * 256 + 3 * (2 * 32 * 256)
                       + 2 * 128 * 256 + 2 * 128 * 256;
  const int prep_blocks = (prep_total + 255) / 256;
  prep_weights<<<prep_blocks, 256, 0, stream>>>(
      c2d_w, c2d_b, bn2d_g, bn2d_b, bn2d_m, bn2d_v,
      wA, wV, qkv_w, proj_w, fc1_w, fc2_w,
      W2pk, bias2, wApk, wVpk, qwpk, pwpk, fc1pk, fc2pk);

  lsga_fused<<<1024, 256, SMEM_BYTES, stream>>>(
      x, c3d_w, c3d_b, bn3d_g, bn3d_b, bn3d_m, bn3d_v,
      pe_ln_g, pe_ln_b, n1_g, n1_b, qkv_b, proj_b, n2_g, n2_b,
      fc1_b, fc2_b, fn_g, fn_b, head_w, head_b,
      W2pk, bias2, wApk, wVpk, qwpk, pwpk, fc1pk, fc2pk,
      (float*)d_out);
}
